// GroupLinear_66348654788888
// MI455X (gfx1250) — matrix operand order repaired and checked by reading, not compiled, not run on hardware
//
#include <hip/hip_runtime.h>

// Problem constants (fixed by the reference).
#define TOK  8192
#define NGRP 8
#define KDIM 1024
#define NDIM 2048

// Tiling.
#define TM 128
#define TN 128
#define TK 32
#define LDSW 40   // ushort elements per LDS row (32 data + 8 pad) = 80 bytes -> bank-conflict-free b128 frags

typedef __attribute__((ext_vector_type(16))) __bf16 v16bf;
typedef __attribute__((ext_vector_type(8)))  float  v8f;

struct Frag32B { uint4 a, b; };   // 32 bytes, same size as v16bf

static __device__ __forceinline__ unsigned short f32_to_bf16_rne(float f) {
    unsigned u = __float_as_uint(f);
    unsigned r = u + 0x7FFFu + ((u >> 16) & 1u);
    return (unsigned short)(r >> 16);
}

static __device__ __forceinline__ void split4(float4 v,
                                              unsigned short* __restrict__ hi,
                                              unsigned short* __restrict__ lo) {
    float f[4] = {v.x, v.y, v.z, v.w};
    unsigned short h[4], l[4];
#pragma unroll
    for (int i = 0; i < 4; ++i) {
        h[i] = f32_to_bf16_rne(f[i]);
        float hf = __uint_as_float((unsigned)h[i] << 16);
        l[i] = f32_to_bf16_rne(f[i] - hf);
    }
    uint2 hp, lp;
    hp.x = (unsigned)h[0] | ((unsigned)h[1] << 16);
    hp.y = (unsigned)h[2] | ((unsigned)h[3] << 16);
    lp.x = (unsigned)l[0] | ((unsigned)l[1] << 16);
    lp.y = (unsigned)l[2] | ((unsigned)l[3] << 16);
    *(uint2*)hi = hp;
    *(uint2*)lo = lp;
}

static __device__ __forceinline__ v16bf ld_frag(const unsigned short* p0,
                                                const unsigned short* p1) {
    Frag32B f;
    f.a = *(const uint4*)p0;
    f.b = *(const uint4*)p1;
    return __builtin_bit_cast(v16bf, f);
}

// CDNA5 async global->LDS copy: 16B per lane, tracked by ASYNCcnt (no VGPR round-trip).
static __device__ __forceinline__ void async_b128(const unsigned short* g, unsigned short* l) {
    unsigned lds = (unsigned)(size_t)l;   // flat->LDS: low 32 bits are the LDS byte address
    asm volatile("global_load_async_to_lds_b128 %0, %1, off"
                 :: "v"(lds), "v"(g) : "memory");
}
static __device__ __forceinline__ void wait_async0() {
    asm volatile("s_wait_asynccnt 0x0" ::: "memory");
}

// ---------------------------------------------------------------------------
// Pre-pass: split f32 into bf16 hi/lo planes (bandwidth-bound, runs once).
// ---------------------------------------------------------------------------
__global__ __launch_bounds__(256)
void split_planes(const float* __restrict__ src,
                  unsigned short* __restrict__ hi,
                  unsigned short* __restrict__ lo, int n4) {
    int i = blockIdx.x * blockDim.x + threadIdx.x;
    if (i >= n4) return;
    float4 v = ((const float4*)src)[i];
    split4(v, hi + 4 * (size_t)i, lo + 4 * (size_t)i);
}

// ---------------------------------------------------------------------------
// Main GEMM: bf16x3 from pre-split planes, async-LDS double buffering.
// ---------------------------------------------------------------------------
__global__ __launch_bounds__(256)
void grouped_gemm_bf16_pre(const unsigned short* __restrict__ xh,
                           const unsigned short* __restrict__ xl,
                           const unsigned short* __restrict__ wh,
                           const unsigned short* __restrict__ wl,
                           const int* __restrict__ offs,
                           float* __restrict__ out) {
    // [buf][hi/lo][row][col] : 2*2*128*40*2 B = 40 KB each, 80 KB total.
    __shared__ __align__(16) unsigned short sA[2][2][TM][LDSW];
    __shared__ __align__(16) unsigned short sW[2][2][TN][LDSW];

    const int g       = blockIdx.z;
    const int g_start = (g == 0) ? 0 : offs[g - 1];
    const int g_end   = offs[g];
    const int rows    = g_end - g_start;
    const int m0      = blockIdx.x * TM;
    if (m0 >= rows) return;
    const int row0 = g_start + m0;
    const int n0   = blockIdx.y * TN;

    const int tid  = threadIdx.x;
    const int lane = tid & 31;
    const int wv   = tid >> 5;
    const int wr   = wv >> 2;       // 0..1 : 64-row band
    const int wc   = wv & 3;        // 0..3 : 32-col band
    const int lr   = lane & 15;
    const int lh   = lane >> 4;

    const size_t wgbase = (size_t)g * NDIM * KDIM;

    // Per-thread staging coordinates: 512 16B-segments per plane, 2 per thread.
    int seg0 = tid, seg1 = tid + 256;
    int r0s = seg0 >> 2, c0s = (seg0 & 3) * 8;
    int r1s = seg1 >> 2, c1s = (seg1 & 3) * 8;
    int ga0 = row0 + r0s; if (ga0 > TOK - 1) ga0 = TOK - 1;   // clipped tail rows: read
    int ga1 = row0 + r1s; if (ga1 > TOK - 1) ga1 = TOK - 1;   // in-bounds, never stored

    v8f acc[4][2];
    const v8f vzero = {0.f, 0.f, 0.f, 0.f, 0.f, 0.f, 0.f, 0.f};
#pragma unroll
    for (int mt = 0; mt < 4; ++mt)
#pragma unroll
        for (int nt = 0; nt < 2; ++nt) acc[mt][nt] = vzero;

    // Stage k-tile kk into buffer b (8 async b128 per thread = 32 KB per block).
    auto stage = [&](int kk, int b) {
        size_t go0 = (size_t)ga0 * KDIM + kk + c0s;
        size_t go1 = (size_t)ga1 * KDIM + kk + c1s;
        size_t wo0 = wgbase + (size_t)(n0 + r0s) * KDIM + kk + c0s;
        size_t wo1 = wgbase + (size_t)(n0 + r1s) * KDIM + kk + c1s;
        async_b128(xh + go0, &sA[b][0][r0s][c0s]);
        async_b128(xl + go0, &sA[b][1][r0s][c0s]);
        async_b128(xh + go1, &sA[b][0][r1s][c1s]);
        async_b128(xl + go1, &sA[b][1][r1s][c1s]);
        async_b128(wh + wo0, &sW[b][0][r0s][c0s]);
        async_b128(wl + wo0, &sW[b][1][r0s][c0s]);
        async_b128(wh + wo1, &sW[b][0][r1s][c1s]);
        async_b128(wl + wo1, &sW[b][1][r1s][c1s]);
    };

    stage(0, 0);
    wait_async0();
    __syncthreads();

    for (int kk = 0; kk < KDIM; kk += TK) {
        const int cur = (kk >> 5) & 1;
        if (kk + TK < KDIM) stage(kk + TK, cur ^ 1);   // prefetch next tile

        // B fragments (contiguous k = 16*lh + s, same order as A).
        v16bf bh[2], bl[2];
#pragma unroll
        for (int nt = 0; nt < 2; ++nt) {
            const unsigned short* pbh = &sW[cur][0][wc * 32 + nt * 16 + lr][lh * 16];
            const unsigned short* pbl = &sW[cur][1][wc * 32 + nt * 16 + lr][lh * 16];
            bh[nt] = ld_frag(pbh, pbh + 8);
            bl[nt] = ld_frag(pbl, pbl + 8);
        }
        // A fragments (contiguous k = 16*lh + s, same order as B) + bf16x3 WMMA.
#pragma unroll
        for (int mt = 0; mt < 4; ++mt) {
            const unsigned short* pah = &sA[cur][0][wr * 64 + mt * 16 + lr][lh * 16];
            const unsigned short* pal = &sA[cur][1][wr * 64 + mt * 16 + lr][lh * 16];
            v16bf ah = ld_frag(pah, pah + 8);
            v16bf al = ld_frag(pal, pal + 8);
#pragma unroll
            for (int nt = 0; nt < 2; ++nt) {
                acc[mt][nt] = __builtin_amdgcn_wmma_f32_16x16x32_bf16(
                    false, ah, false, bh[nt], (short)0, acc[mt][nt], false, false);
                acc[mt][nt] = __builtin_amdgcn_wmma_f32_16x16x32_bf16(
                    false, al, false, bh[nt], (short)0, acc[mt][nt], false, false);
                acc[mt][nt] = __builtin_amdgcn_wmma_f32_16x16x32_bf16(
                    false, ah, false, bl[nt], (short)0, acc[mt][nt], false, false);
            }
        }

        wait_async0();       // next buffer filled
        __syncthreads();     // all waves done reading cur (dscnt waited before wmma)
    }

#pragma unroll
    for (int mt = 0; mt < 4; ++mt) {
#pragma unroll
        for (int nt = 0; nt < 2; ++nt) {
            int rbase = row0 + wr * 64 + mt * 16 + lh * 8;
            int cn    = n0 + wc * 32 + nt * 16 + lr;
#pragma unroll
            for (int i = 0; i < 8; ++i) {
                int gr = rbase + i;
                if (gr < g_end) out[(size_t)gr * NDIM + cn] = acc[mt][nt][i];
            }
        }
    }
}

// ---------------------------------------------------------------------------
// Fallback: fused conversion kernel (round-1 version), used if ws too small.
// ---------------------------------------------------------------------------
__global__ __launch_bounds__(256)
void grouped_gemm_bf16x3(const float* __restrict__ x,
                         const float* __restrict__ w,
                         const int*   __restrict__ offs,
                         float*       __restrict__ out) {
    __shared__ __align__(16) unsigned short sAh[TM][LDSW];
    __shared__ __align__(16) unsigned short sAl[TM][LDSW];
    __shared__ __align__(16) unsigned short sWh[TN][LDSW];
    __shared__ __align__(16) unsigned short sWl[TN][LDSW];

    const int g       = blockIdx.z;
    const int g_start = (g == 0) ? 0 : offs[g - 1];
    const int g_end   = offs[g];
    const int rows    = g_end - g_start;
    const int m0      = blockIdx.x * TM;
    if (m0 >= rows) return;
    const int row0 = g_start + m0;
    const int n0   = blockIdx.y * TN;

    const int tid  = threadIdx.x;
    const int lane = tid & 31;
    const int wv   = tid >> 5;
    const int wr   = wv >> 2;
    const int wc   = wv & 3;
    const int lr   = lane & 15;
    const int lh   = lane >> 4;

    v8f acc[4][2];
    const v8f vzero = {0.f, 0.f, 0.f, 0.f, 0.f, 0.f, 0.f, 0.f};
#pragma unroll
    for (int mt = 0; mt < 4; ++mt)
#pragma unroll
        for (int nt = 0; nt < 2; ++nt) acc[mt][nt] = vzero;

    const float* wg = w + (size_t)g * NDIM * KDIM;

    for (int kk = 0; kk < KDIM; kk += TK) {
        __syncthreads();
#pragma unroll
        for (int it = 0; it < 4; ++it) {
            int idx = tid + it * 256;
            int r   = idx >> 3;
            int c   = (idx & 7) << 2;
            int gr  = row0 + r;
            if (gr > TOK - 1) gr = TOK - 1;
            float4 v = *(const float4*)(x + (size_t)gr * KDIM + kk + c);
            split4(v, &sAh[r][c], &sAl[r][c]);
        }
#pragma unroll
        for (int it = 0; it < 4; ++it) {
            int idx = tid + it * 256;
            int r   = idx >> 3;
            int c   = (idx & 7) << 2;
            float4 v = *(const float4*)(wg + (size_t)(n0 + r) * KDIM + kk + c);
            split4(v, &sWh[r][c], &sWl[r][c]);
        }
        __syncthreads();

        v16bf bh[2], bl[2];
#pragma unroll
        for (int nt = 0; nt < 2; ++nt) {
            const unsigned short* pbh = &sWh[wc * 32 + nt * 16 + lr][lh * 16];
            const unsigned short* pbl = &sWl[wc * 32 + nt * 16 + lr][lh * 16];
            bh[nt] = ld_frag(pbh, pbh + 8);
            bl[nt] = ld_frag(pbl, pbl + 8);
        }
#pragma unroll
        for (int mt = 0; mt < 4; ++mt) {
            const unsigned short* pah = &sAh[wr * 64 + mt * 16 + lr][lh * 16];
            const unsigned short* pal = &sAl[wr * 64 + mt * 16 + lr][lh * 16];
            v16bf ah = ld_frag(pah, pah + 8);
            v16bf al = ld_frag(pal, pal + 8);
#pragma unroll
            for (int nt = 0; nt < 2; ++nt) {
                acc[mt][nt] = __builtin_amdgcn_wmma_f32_16x16x32_bf16(
                    false, ah, false, bh[nt], (short)0, acc[mt][nt], false, false);
                acc[mt][nt] = __builtin_amdgcn_wmma_f32_16x16x32_bf16(
                    false, al, false, bh[nt], (short)0, acc[mt][nt], false, false);
                acc[mt][nt] = __builtin_amdgcn_wmma_f32_16x16x32_bf16(
                    false, ah, false, bl[nt], (short)0, acc[mt][nt], false, false);
            }
        }
    }

#pragma unroll
    for (int mt = 0; mt < 4; ++mt) {
#pragma unroll
        for (int nt = 0; nt < 2; ++nt) {
            int rbase = row0 + wr * 64 + mt * 16 + lh * 8;
            int cn    = n0 + wc * 32 + nt * 16 + lr;
#pragma unroll
            for (int i = 0; i < 8; ++i) {
                int gr = rbase + i;
                if (gr < g_end) out[(size_t)gr * NDIM + cn] = acc[mt][nt][i];
            }
        }
    }
}

extern "C" void kernel_launch(void* const* d_in, const int* in_sizes, int n_in,
                              void* d_out, int out_size, void* d_ws, size_t ws_size,
                              hipStream_t stream) {
    (void)in_sizes; (void)n_in; (void)out_size;
    const float* x    = (const float*)d_in[0];
    const float* w    = (const float*)d_in[1];
    const int*   offs = (const int*)d_in[2];
    float*       out  = (float*)d_out;

    const size_t xElems = (size_t)TOK * KDIM;          // 8.39 M
    const size_t wElems = (size_t)NGRP * NDIM * KDIM;  // 16.78 M
    const size_t need   = 2 * (xElems + wElems) * sizeof(unsigned short); // ~96 MB

    dim3 grid(TOK / TM, NDIM / TN, NGRP);
    dim3 block(256);

    if (ws_size >= need) {
        unsigned short* xh = (unsigned short*)d_ws;
        unsigned short* xl = xh + xElems;
        unsigned short* wh = xl + xElems;
        unsigned short* wl = wh + wElems;
        int xn4 = (int)(xElems / 4), wn4 = (int)(wElems / 4);
        hipLaunchKernelGGL(split_planes, dim3((xn4 + 255) / 256), block, 0, stream, x, xh, xl, xn4);
        hipLaunchKernelGGL(split_planes, dim3((wn4 + 255) / 256), block, 0, stream, w, wh, wl, wn4);
        hipLaunchKernelGGL(grouped_gemm_bf16_pre, grid, block, 0, stream, xh, xl, wh, wl, offs, out);
    } else {
        hipLaunchKernelGGL(grouped_gemm_bf16x3, grid, block, 0, stream, x, w, offs, out);
    }
}
